// LogicLayer_58901181497985
// MI455X (gfx1250) — compile-verified, matrix operand order checked by reading, not run on hardware
//
#include <hip/hip_runtime.h>

typedef __attribute__((ext_vector_type(16))) _Float16 v16h;
typedef __attribute__((ext_vector_type(8)))  float    v8f;

#define NB   32
#define NN   128

__device__ __forceinline__ v8f wmma32f16(v16h a, v16h b, v8f c) {
  // D = A(16x32 f16) x B(32x16 f16) + C(16x16 f32)
  return __builtin_amdgcn_wmma_f32_16x16x32_f16(false, a, false, b, (short)0, c,
                                                false, false);
}

__device__ __forceinline__ int kmap(int j, int half) {
  // ISA 16-bit A/B fragment K index for element j of a lane (half = lane/16)
  return j + 8 * half + ((j < 8) ? 0 : 8);
}

__device__ __forceinline__ float sigmoidf_fast(float x) {
  return 1.0f / (1.0f + __expf(-x));
}

// ---------------------------------------------------------------------------
// Reducer(2): for each (b,i,c): max/min over j != i of x2[b,i,j,c], written
// interleaved [ex0,fa0,ex1,fa1,...] to match reference stack+reshape.
// Masked slots contribute 0.0 (max) / 1.0 (min) exactly like jnp.where.
// ---------------------------------------------------------------------------
__global__ void __launch_bounds__(256) red2_kernel(const float* __restrict__ x2,
                                                   float* __restrict__ red2) {
  int idx = blockIdx.x * blockDim.x + threadIdx.x;   // B*N*C2 = 65536
  int c  = idx & 15;
  int bi = idx >> 4;          // b*128 + i
  int i  = bi & 127;
  const float* row = x2 + (size_t)bi * NN * 16 + c;
  float mx = 0.0f, mn = 1.0f;
  for (int j = 0; j < NN; ++j) {
    if (j == i) continue;
    float v = row[(size_t)j * 16];
    mx = fmaxf(mx, v);
    mn = fminf(mn, v);
  }
  red2[bi * 32 + 2 * c]     = mx;
  red2[bi * 32 + 2 * c + 1] = mn;
}

// ---------------------------------------------------------------------------
// Group 0 (nullary): 32 rows, 48 -> 64 -> 16 scalar MLP (trivial work).
// ---------------------------------------------------------------------------
__global__ void __launch_bounds__(64) g0_kernel(const float* __restrict__ x0,
                                                const float* __restrict__ x1,
                                                const float* __restrict__ W1,
                                                const float* __restrict__ b1,
                                                const float* __restrict__ W2,
                                                const float* __restrict__ b2,
                                                float* __restrict__ out0) {
  __shared__ float f0[48];
  __shared__ float h[64];
  const int b = blockIdx.x;
  const int t = threadIdx.x;
  if (t < 16) f0[t] = x0[b * 16 + t];
  if (t >= 16 && t < 32) {
    int c = t - 16;
    float mx = -1e30f, mn = 1e30f;
    for (int n = 0; n < NN; ++n) {
      float v = x1[(size_t)(b * NN + n) * 16 + c];
      mx = fmaxf(mx, v);
      mn = fminf(mn, v);
    }
    f0[16 + 2 * c]     = mx;   // exists
    f0[16 + 2 * c + 1] = mn;   // forall
  }
  __syncthreads();
  float acc = b1[t];
  for (int k = 0; k < 48; ++k) acc += f0[k] * W1[k * 64 + t];
  h[t] = fmaxf(acc, 0.0f);
  __syncthreads();
  if (t < 16) {
    float a = b2[t];
    for (int k = 0; k < 64; ++k) a += h[k] * W2[k * 16 + t];
    out0[b * 16 + t] = sigmoidf_fast(a);
  }
}

// ---------------------------------------------------------------------------
// Group 1 (unary): 4096 rows x (64 -> 64 -> 16) via WMMA. One 16-row tile per
// wave. Features per row rr=(b*128+i): [x0[b] | x1[rr] | red2[rr] (32)].
// ---------------------------------------------------------------------------
__global__ void __launch_bounds__(256) g1_wmma_kernel(
    const float* __restrict__ x0, const float* __restrict__ x1,
    const float* __restrict__ red2,
    const float* __restrict__ W1, const float* __restrict__ b1,
    const float* __restrict__ W2, const float* __restrict__ b2,
    float* __restrict__ out1, int ntiles) {
  __shared__ _Float16 hbuf_all[8][16 * 64];
  const int lane = threadIdx.x & 31;
  const int wave = threadIdx.x >> 5;
  const int half = lane >> 4;
  const int m16  = lane & 15;
  const int wid  = blockIdx.x * 8 + wave;
  const int nwaves = gridDim.x * 8;
  _Float16* hbuf = hbuf_all[wave];

  // B-matrix weight fragments (one-time gather, L2-resident weights)
  v16h w1f[2][4];
#pragma unroll
  for (int kc = 0; kc < 2; ++kc)
#pragma unroll
    for (int ng = 0; ng < 4; ++ng)
#pragma unroll
      for (int j = 0; j < 16; ++j)
        w1f[kc][ng][j] = (_Float16)W1[(kmap(j, half) + 32 * kc) * 64 + 16 * ng + m16];
  v16h w2f[2];
#pragma unroll
  for (int kc = 0; kc < 2; ++kc)
#pragma unroll
    for (int j = 0; j < 16; ++j)
      w2f[kc][j] = (_Float16)W2[(kmap(j, half) + 32 * kc) * 16 + m16];
  float bias1[4];
#pragma unroll
  for (int ng = 0; ng < 4; ++ng) bias1[ng] = b1[16 * ng + m16];
  const float bias2 = b2[m16];

  for (int t = wid; t < ntiles; t += nwaves) {
    const int rr = 16 * t + m16;            // this lane's row = b*128 + i
    const int b  = rr >> 7;
    const float* x0b  = x0   + b * 16;
    const float* x1r  = x1   + (size_t)rr * 16;
    const float* r2r  = red2 + (size_t)rr * 32;

    v16h a0, a1;   // K = 0..31 and 32..63 (contiguous 8-float runs)
#pragma unroll
    for (int j = 0; j < 8; ++j) {
      a0[j]     = (_Float16)x0b[j + 8 * half];
      a0[j + 8] = (_Float16)x1r[j + 8 * half];
      a1[j]     = (_Float16)r2r[j + 8 * half];
      a1[j + 8] = (_Float16)r2r[16 + j + 8 * half];
    }

    v8f acc[4];
#pragma unroll
    for (int ng = 0; ng < 4; ++ng) {
      v8f z = {};
      z = wmma32f16(a0, w1f[0][ng], z);
      z = wmma32f16(a1, w1f[1][ng], z);
      acc[ng] = z;
    }

#pragma unroll
    for (int ng = 0; ng < 4; ++ng)
#pragma unroll
      for (int r = 0; r < 8; ++r) {
        float v = acc[ng][r] + bias1[ng];
        hbuf[(r + 8 * half) * 64 + 16 * ng + m16] = (_Float16)fmaxf(v, 0.0f);
      }
    __syncthreads();

    v16h h0, h1;
#pragma unroll
    for (int j = 0; j < 8; ++j) {
      h0[j]     = hbuf[m16 * 64 + j + 8 * half];
      h0[j + 8] = hbuf[m16 * 64 + 16 + j + 8 * half];
      h1[j]     = hbuf[m16 * 64 + 32 + j + 8 * half];
      h1[j + 8] = hbuf[m16 * 64 + 48 + j + 8 * half];
    }
    v8f d = {};
    d = wmma32f16(h0, w2f[0], d);
    d = wmma32f16(h1, w2f[1], d);

    float* orow = out1 + (size_t)(16 * t) * 16;
#pragma unroll
    for (int r = 0; r < 8; ++r)
      orow[(r + 8 * half) * 16 + m16] = sigmoidf_fast(d[r] + bias2);
    __syncthreads();
  }
}

// ---------------------------------------------------------------------------
// Group 2 (binary): 524288 rows x (64 -> 64 -> 16) via WMMA. One 16-row tile
// per wave (fixed (b,i), 16 consecutive j). Feature per row (b,i,j):
// [x1[b,i] | x2[b,i,j] | x1[b,j] | x2[b,j,i]].
// ---------------------------------------------------------------------------
__global__ void __launch_bounds__(256) g2_wmma_kernel(
    const float* __restrict__ x1, const float* __restrict__ x2,
    const float* __restrict__ W1, const float* __restrict__ b1,
    const float* __restrict__ W2, const float* __restrict__ b2,
    float* __restrict__ out2, int ntiles) {
  __shared__ _Float16 hbuf_all[8][16 * 64];
  const int lane = threadIdx.x & 31;
  const int wave = threadIdx.x >> 5;
  const int half = lane >> 4;
  const int m16  = lane & 15;
  const int wid  = blockIdx.x * 8 + wave;
  const int nwaves = gridDim.x * 8;
  _Float16* hbuf = hbuf_all[wave];

  v16h w1f[2][4];
#pragma unroll
  for (int kc = 0; kc < 2; ++kc)
#pragma unroll
    for (int ng = 0; ng < 4; ++ng)
#pragma unroll
      for (int j = 0; j < 16; ++j)
        w1f[kc][ng][j] = (_Float16)W1[(kmap(j, half) + 32 * kc) * 64 + 16 * ng + m16];
  v16h w2f[2];
#pragma unroll
  for (int kc = 0; kc < 2; ++kc)
#pragma unroll
    for (int j = 0; j < 16; ++j)
      w2f[kc][j] = (_Float16)W2[(kmap(j, half) + 32 * kc) * 16 + m16];
  float bias1[4];
#pragma unroll
  for (int ng = 0; ng < 4; ++ng) bias1[ng] = b1[16 * ng + m16];
  const float bias2 = b2[m16];

  for (int t = wid; t < ntiles; t += nwaves) {
    const int b  = t >> 10;            // 1024 tiles per batch (128*8)
    const int i  = (t & 1023) >> 3;
    const int j0 = (t & 7) << 4;
    const int jr = j0 + m16;           // this lane's j row
    const float* x1i  = x1 + (size_t)(b * NN + i) * 16;
    const float* x1j  = x1 + (size_t)(b * NN + jr) * 16;
    const float* x2ij = x2 + ((size_t)(b * NN + i) * NN + jr) * 16;
    const float* x2ji = x2 + ((size_t)(b * NN + jr) * NN + i) * 16;

    // prefetch next tile's x2 rows into cache
    int tn = t + nwaves;
    if (tn < ntiles) {
      int bn = tn >> 10, in_ = (tn & 1023) >> 3, j0n = (tn & 7) << 4;
      __builtin_prefetch(x2 + ((size_t)(bn * NN + in_) * NN + j0n + m16) * 16, 0, 1);
    }

    v16h a0, a1;   // K runs are contiguous 8-float chunks -> b128 loads
#pragma unroll
    for (int j = 0; j < 8; ++j) {
      a0[j]     = (_Float16)x1i [j + 8 * half];
      a0[j + 8] = (_Float16)x2ij[j + 8 * half];
      a1[j]     = (_Float16)x1j [j + 8 * half];
      a1[j + 8] = (_Float16)x2ji[j + 8 * half];
    }

    v8f acc[4];
#pragma unroll
    for (int ng = 0; ng < 4; ++ng) {
      v8f z = {};
      z = wmma32f16(a0, w1f[0][ng], z);
      z = wmma32f16(a1, w1f[1][ng], z);
      acc[ng] = z;
    }

#pragma unroll
    for (int ng = 0; ng < 4; ++ng)
#pragma unroll
      for (int r = 0; r < 8; ++r) {
        float v = acc[ng][r] + bias1[ng];
        hbuf[(r + 8 * half) * 64 + 16 * ng + m16] = (_Float16)fmaxf(v, 0.0f);
      }
    __syncthreads();

    v16h h0, h1;
#pragma unroll
    for (int j = 0; j < 8; ++j) {
      h0[j]     = hbuf[m16 * 64 + j + 8 * half];
      h0[j + 8] = hbuf[m16 * 64 + 16 + j + 8 * half];
      h1[j]     = hbuf[m16 * 64 + 32 + j + 8 * half];
      h1[j + 8] = hbuf[m16 * 64 + 48 + j + 8 * half];
    }
    v8f d = {};
    d = wmma32f16(h0, w2f[0], d);
    d = wmma32f16(h1, w2f[1], d);

    float* orow = out2 + (size_t)(16 * t) * 16;
#pragma unroll
    for (int r = 0; r < 8; ++r)
      orow[(r + 8 * half) * 16 + m16] = sigmoidf_fast(d[r] + bias2);
    __syncthreads();
  }
}

// ---------------------------------------------------------------------------
extern "C" void kernel_launch(void* const* d_in, const int* in_sizes, int n_in,
                              void* d_out, int out_size, void* d_ws, size_t ws_size,
                              hipStream_t stream) {
  (void)in_sizes; (void)n_in; (void)out_size; (void)ws_size;
  const float* x0   = (const float*)d_in[0];
  const float* x1   = (const float*)d_in[1];
  const float* x2   = (const float*)d_in[2];
  const float* W1_0 = (const float*)d_in[3];
  const float* b1_0 = (const float*)d_in[4];
  const float* W2_0 = (const float*)d_in[5];
  const float* b2_0 = (const float*)d_in[6];
  const float* W1_1 = (const float*)d_in[7];
  const float* b1_1 = (const float*)d_in[8];
  const float* W2_1 = (const float*)d_in[9];
  const float* b2_1 = (const float*)d_in[10];
  const float* W1_2 = (const float*)d_in[11];
  const float* b1_2 = (const float*)d_in[12];
  const float* W2_2 = (const float*)d_in[13];
  const float* b2_2 = (const float*)d_in[14];

  float* out0 = (float*)d_out;                  // [32,16]
  float* out1 = out0 + NB * 16;                 // [32,128,16]
  float* out2 = out1 + NB * NN * 16;            // [32,128,128,16]
  float* red2 = (float*)d_ws;                   // [32,128,32] = 512 KB

  // Reducer(2) for group-1 features
  red2_kernel<<<dim3(256), dim3(256), 0, stream>>>(x2, red2);
  // Group 0 (tiny scalar MLP, includes Reducer(1))
  g0_kernel<<<dim3(NB), dim3(64), 0, stream>>>(x0, x1, W1_0, b1_0, W2_0, b2_0, out0);
  // Group 1: 256 tiles, 256 waves -> 1 tile/wave (uniform loop)
  g1_wmma_kernel<<<dim3(32), dim3(256), 0, stream>>>(x0, x1, red2,
                                                     W1_1, b1_1, W2_1, b2_1,
                                                     out1, 256);
  // Group 2: 32768 tiles, 4096 waves -> 8 tiles/wave (uniform loop)
  g2_wmma_kernel<<<dim3(512), dim3(256), 0, stream>>>(x1, x2,
                                                      W1_2, b1_2, W2_2, b2_2,
                                                      out2, 32768);
}
